// CoreRNNFW_17016660427341
// MI455X (gfx1250) — matrix operand order repaired and checked by reading, not compiled, not run on hardware
//
#include <hip/hip_runtime.h>

// ---------------- problem constants ----------------
constexpr int T_STEPS = 48;
constexpr int B_TOT   = 96;
constexpr int DGc     = 192;
constexpr int DHc     = 384;
constexpr int S_IT    = 3;
constexpr float LAMc  = 0.9f;
constexpr float ETAc  = 0.5f;
constexpr float LNEPS = 1e-5f;

// ---------------- mapping constants ----------------
constexpr int GROUPS  = 6;            // 6 batch-groups of 16  (96 = 6*16)
constexpr int NSLICE  = 8;            // DH sliced across 8 WGs per group
constexpr int SLICE   = DHc / NSLICE; // 48 rows of A / Ah per WG
constexpr int NWG     = GROUPS * NSLICE; // 48 workgroups
constexpr int BLK     = 256;          // 8 waves (wave32)
constexpr int NT      = DHc / 16;     // 24 N-tiles of 16
constexpr int KT_H    = DHc / 32;     // 12 K-steps (bf16 wmma K=32)
constexpr int KT_G    = DGc / 32;     // 6
constexpr int TILES_PER_WG = SLICE / 16;       // 3 wmma N-tiles per WG
constexpr int PAIRS_PER_TH = (16 * SLICE) / BLK; // 3 (b,row) pairs per thread

// ---------------- workspace layout (in floats) ----------------
constexpr size_t OFF_A   = 0;                                   // A: [96][384][384] f32 (56.6 MB, L2-resident)
constexpr size_t SZ_A    = (size_t)B_TOT * DHc * DHc;
constexpr size_t OFF_HB  = OFF_A + SZ_A;                        // h_base exchange: [6][16][384]
constexpr size_t SZ_HB   = (size_t)GROUPS * 16 * DHc;
constexpr size_t OFF_AH  = OFF_HB + SZ_HB;                      // Ah exchange, double-buffered: [2][6][16][384]
constexpr size_t SZ_AH   = (size_t)2 * GROUPS * 16 * DHc;
constexpr size_t OFF_BPH = OFF_AH + SZ_AH;                      // packed bf16 W_h^T operands (12*24*512 bf16)
constexpr size_t SZ_BPH  = (size_t)KT_H * NT * 512 / 2;         // in floats
constexpr size_t OFF_BPG = OFF_BPH + SZ_BPH;                    // packed bf16 W_g^T operands (6*24*512 bf16)
constexpr size_t SZ_BPG  = (size_t)KT_G * NT * 512 / 2;
constexpr size_t OFF_BAR = OFF_BPG + SZ_BPG;                    // spin-barrier counters [6][256] u32
constexpr size_t SZ_BAR  = (size_t)GROUPS * 256;
constexpr size_t WS_FLOATS = OFF_BAR + SZ_BAR;                  // ~57.5 MB total

// ---------------- types ----------------
typedef __attribute__((ext_vector_type(16))) __bf16 v16bf;
typedef __attribute__((ext_vector_type(8)))  float  v8f;
typedef __attribute__((ext_vector_type(4)))  float  f32x4;
typedef __attribute__((ext_vector_type(4)))  unsigned int u32x4;
union ABu { v16bf v; u32x4 q[2]; };

__device__ __forceinline__ unsigned short f2bf(float f) {
  unsigned u = __builtin_bit_cast(unsigned, f);
  unsigned r = u + 0x7FFFu + ((u >> 16) & 1u);   // round-to-nearest-even
  return (unsigned short)(r >> 16);
}

// agent-scope split barrier across the NSLICE workgroups of one batch-group.
// Safe: 48 WGs are trivially co-resident on MI455X.
// __threadfence() = device-scope fence (store completion + L0 invalidate).
__device__ __forceinline__ void group_barrier(unsigned* c) {
  __threadfence();                      // release: make our slice stores visible
  __syncthreads();
  if (threadIdx.x == 0) {
    __hip_atomic_fetch_add(c, 1u, __ATOMIC_RELEASE, __HIP_MEMORY_SCOPE_AGENT);
    while (__hip_atomic_load(c, __ATOMIC_ACQUIRE, __HIP_MEMORY_SCOPE_AGENT) < (unsigned)NSLICE) {
      __builtin_amdgcn_s_sleep(2);
    }
  }
  __syncthreads();
  __threadfence();                      // acquire: don't read stale lines
}

// ---------------- kernels ----------------
__global__ void fw_rnn_zero(float* ws, long n) {
  long i = (long)blockIdx.x * blockDim.x + threadIdx.x;
  if (i < n) ws[i] = 0.0f;
}

// Pack W_h^T / W_g^T into bf16 WMMA-B operand order:
// tile (kt,nt): lane L holds column n = nt*16 + (L&15), K = kt*32 + (L>=16?16:0) + j, j=0..15.
__global__ void fw_rnn_pack(const float* __restrict__ Wh, const float* __restrict__ Wg,
                            float* __restrict__ ws) {
  unsigned short* BPH = (unsigned short*)(ws + OFF_BPH);
  unsigned short* BPG = (unsigned short*)(ws + OFF_BPG);
  const int NH = KT_H * NT * 512;   // 147456
  const int NG = KT_G * NT * 512;   // 73728
  int idx = blockIdx.x * blockDim.x + threadIdx.x;
  if (idx < NH) {
    int j = idx & 15, ln = (idx >> 4) & 31, tile = idx >> 9;
    int nt = tile % NT, kt = tile / NT;
    int k = kt * 32 + ((ln >= 16) ? 16 : 0) + j;
    int n = nt * 16 + (ln & 15);
    BPH[idx] = f2bf(Wh[n * DHc + k]);          // B[k][n] = W_h[n][k]
  } else if (idx < NH + NG) {
    int i2 = idx - NH;
    int j = i2 & 15, ln = (i2 >> 4) & 31, tile = i2 >> 9;
    int nt = tile % NT, kt = tile / NT;
    int k = kt * 32 + ((ln >= 16) ? 16 : 0) + j;
    int n = nt * 16 + (ln & 15);
    BPG[i2] = f2bf(Wg[n * DGc + k]);           // B[k][n] = W_g[n][k]
  }
}

__global__ __launch_bounds__(BLK)
void fw_rnn_main(const float* __restrict__ z,
                 const float* __restrict__ bh,
                 const float* __restrict__ lng_g,
                 const float* __restrict__ lnb_g,
                 const float* __restrict__ alpha,
                 float* __restrict__ out,
                 float* __restrict__ ws) {
  __shared__ float hs[16][DHc];                    // current h_s (f32)
  __shared__ float hbase[16][DHc];                 // full h_base for this step
  __shared__ unsigned short hbf[16][DHc + 16];     // bf16 copy of h (WMMA A operand)
  __shared__ unsigned short zbf[16][DGc + 16];     // bf16 z tile
  __shared__ float s_lng[DHc], s_lnb[DHc], s_bh[DHc];

  const int tid  = threadIdx.x;
  const int lane = tid & 31;
  const int wv   = tid >> 5;
  const int g    = blockIdx.x / NSLICE;            // batch group (16 rows)
  const int sl   = blockIdx.x % NSLICE;            // DH slice (48 rows)

  float* A  = ws + OFF_A;
  float* HB = ws + OFF_HB + (size_t)g * 16 * DHc;
  float* AH = ws + OFF_AH;
  const unsigned short* BPH = (const unsigned short*)(ws + OFF_BPH);
  const unsigned short* BPG = (const unsigned short*)(ws + OFF_BPG);
  unsigned* BAR = (unsigned*)(ws + OFF_BAR) + g * 256;

  for (int i = tid; i < DHc; i += BLK) { s_lng[i] = lng_g[i]; s_lnb[i] = lnb_g[i]; s_bh[i] = bh[i]; }
  for (int i = tid; i < 16 * (DHc + 16); i += BLK) ((unsigned short*)hbf)[i] = 0;

  // curvature k from alpha_fw (softplus gate)
  const float af = alpha[0];
  const float kk = (af >= 0.0f) ? (1.0f + log1pf(expf(af)))
                                : (1.0f / (1.0f + log1pf(expf(-af))));
  __syncthreads();

  int pc = 0;      // barrier phase counter
  int ahp = 0;     // Ah buffer parity counter

  for (int t = 0; t < T_STEPS; t++) {
    const bool fin = (t == T_STEPS - 1);

    // ---- stage z_t tile into LDS as bf16 ----
    const float* zt = z + ((size_t)t * B_TOT + g * 16) * DGc;
    #pragma unroll
    for (int e = 0; e < (16 * DGc) / BLK; e++) {
      int li = e * BLK + tid; int b = li / DGc, d = li - b * DGc;
      zbf[b][d] = f2bf(zt[b * DGc + d]);
    }
    __syncthreads();

    // ---- phase 0: h_base slice via WMMA (waves 0..TILES_PER_WG-1) ----
    if (wv < TILES_PER_WG) {
      const int nt = sl * TILES_PER_WG + wv;       // global 16-col tile
      const int m = lane & 15;
      const int hiK = (lane >= 16) ? 16 : 0;       // A-operand K base (contiguous k = 16h + s, as packed B)
      v8f c = {};
      for (int kt = 0; kt < KT_H; kt++) {
        ABu a, b_;
        a.q[0] = *(const u32x4*)&hbf[m][kt * 32 + hiK];
        a.q[1] = *(const u32x4*)&hbf[m][kt * 32 + hiK + 8];
        const u32x4* bp = (const u32x4*)(BPH + ((size_t)(kt * NT + nt) * 32 + lane) * 16);
        b_.q[0] = bp[0]; b_.q[1] = bp[1];
        c = __builtin_amdgcn_wmma_f32_16x16x32_bf16(false, a.v, false, b_.v,
                                                    (short)0, c, false, false);
      }
      for (int kt = 0; kt < KT_G; kt++) {
        ABu a, b_;
        a.q[0] = *(const u32x4*)&zbf[m][kt * 32 + hiK];
        a.q[1] = *(const u32x4*)&zbf[m][kt * 32 + hiK + 8];
        const u32x4* bp = (const u32x4*)(BPG + ((size_t)(kt * NT + nt) * 32 + lane) * 16);
        b_.q[0] = bp[0]; b_.q[1] = bp[1];
        c = __builtin_amdgcn_wmma_f32_16x16x32_bf16(false, a.v, false, b_.v,
                                                    (short)0, c, false, false);
      }
      // C/D layout: VGPR j -> M=j (lanes 0-15) / M=j+8 (lanes 16-31), N = lane&15
      const int n  = nt * 16 + (lane & 15);
      const int mb = (lane >= 16) ? 8 : 0;
      #pragma unroll
      for (int j = 0; j < 8; j++) HB[(size_t)(mb + j) * DHc + n] = c[j] + s_bh[n];
    }
    group_barrier(BAR + pc); pc++;

    // ---- LN(h_base) -> h_s (one wave per 2 batch rows; wave32 shfl reductions) ----
    for (int rep = 0; rep < 2; rep++) {
      const int bi = wv * 2 + rep;
      float uv[12]; float sum = 0.f, sq = 0.f;
      #pragma unroll
      for (int e = 0; e < 12; e++) {
        int idx = e * 32 + lane;
        float u = HB[(size_t)bi * DHc + idx];
        hbase[bi][idx] = u; uv[e] = u; sum += u; sq += u * u;
      }
      for (int m = 1; m < 32; m <<= 1) { sum += __shfl_xor(sum, m, 32); sq += __shfl_xor(sq, m, 32); }
      float mean = sum * (1.0f / DHc);
      float var  = sq * (1.0f / DHc) - mean * mean;
      float rs   = rsqrtf(var + LNEPS);
      #pragma unroll
      for (int e = 0; e < 12; e++) {
        int idx = e * 32 + lane;
        float v = (uv[e] - mean) * rs * s_lng[idx] + s_lnb[idx];
        v = fmaxf(v, 0.0f);
        hs[bi][idx] = v; hbf[bi][idx] = f2bf(v);
      }
    }
    __syncthreads();

    // ---- inner settling loop ----
    for (int it = 0; it < S_IT; it++) {
      const int par = ahp & 1; ahp++;
      float* AHp = AH + ((size_t)par * GROUPS + g) * 16 * DHc;

      // matvec: Ah[b][r] = A[b][r][:] . h_s[b][:]  over this WG's 48 rows
      #pragma unroll
      for (int q = 0; q < PAIRS_PER_TH; q++) {
        int p = tid + BLK * q;
        int b = p / SLICE, rr = p - b * SLICE, r = sl * SLICE + rr;
        const f32x4* Ar = (const f32x4*)(A + ((size_t)(g * 16 + b) * DHc + r) * DHc);
        float acc = 0.f;
        for (int kx = 0; kx < DHc / 4; kx++) {
          f32x4 av = Ar[kx];
          const float* hp = &hs[b][kx * 4];
          acc += av.x * hp[0] + av.y * hp[1] + av.z * hp[2] + av.w * hp[3];
        }
        AHp[(size_t)b * DHc + r] = acc;
      }
      group_barrier(BAR + pc); pc++;

      // gate + LN update (wave per 2 batch rows, all local reductions)
      for (int rep = 0; rep < 2; rep++) {
        const int bi = wv * 2 + rep;
        float av[12]; float dot = 0.f, asq = 0.f, hsq = 0.f;
        #pragma unroll
        for (int e = 0; e < 12; e++) {
          int idx = e * 32 + lane;
          float Av = AHp[(size_t)bi * DHc + idx];
          float hv = hs[bi][idx];
          av[e] = Av; dot += hv * Av; asq += Av * Av; hsq += hv * hv;
        }
        for (int m = 1; m < 32; m <<= 1) {
          dot += __shfl_xor(dot, m, 32);
          asq += __shfl_xor(asq, m, 32);
          hsq += __shfl_xor(hsq, m, 32);
        }
        float wB, wA;
        if (fin) { wB = 1.0f; wA = 1.0f; }           // query step: h_base + Ah, no gate
        else {
          float n1 = fmaxf(sqrtf(hsq), 1e-6f);
          float n2 = fmaxf(sqrtf(asq), 1e-6f);
          float R  = fminf(fmaxf(dot / (n1 * n2), 0.0f), 1.0f);
          float om = 1.0f - R;
          float pw = (om > 0.0f) ? __powf(om, kk) : 0.0f;
          float a  = 1.0f - pw;
          wB = 1.0f - a * a; wA = a;
        }
        float uv[12]; float sum = 0.f, sq = 0.f;
        #pragma unroll
        for (int e = 0; e < 12; e++) {
          int idx = e * 32 + lane;
          float u = wB * hbase[bi][idx] + wA * av[e];
          uv[e] = u; sum += u; sq += u * u;
        }
        for (int m = 1; m < 32; m <<= 1) { sum += __shfl_xor(sum, m, 32); sq += __shfl_xor(sq, m, 32); }
        float mean = sum * (1.0f / DHc);
        float var  = sq * (1.0f / DHc) - mean * mean;
        float rs   = rsqrtf(var + LNEPS);
        #pragma unroll
        for (int e = 0; e < 12; e++) {
          int idx = e * 32 + lane;
          float v = (uv[e] - mean) * rs * s_lng[idx] + s_lnb[idx];
          v = fmaxf(v, 0.0f);
          hs[bi][idx] = v; hbf[bi][idx] = f2bf(v);
        }
      }
      __syncthreads();
    }

    // ---- Hebbian update of this WG's 48 rows of A (local; no cross-WG dep) ----
    if (!fin) {
      #pragma unroll
      for (int q = 0; q < PAIRS_PER_TH; q++) {
        int p = tid + BLK * q;
        int b = p / SLICE, rr = p - b * SLICE, r = sl * SLICE + rr;
        f32x4* Ar = (f32x4*)(A + ((size_t)(g * 16 + b) * DHc + r) * DHc);
        float hr = ETAc * hs[b][r];
        for (int kx = 0; kx < DHc / 4; kx++) {
          f32x4 av = Ar[kx];
          const float* hp = &hs[b][kx * 4];
          av.x = LAMc * av.x + hr * hp[0];
          av.y = LAMc * av.y + hr * hp[1];
          av.z = LAMc * av.z + hr * hp[2];
          av.w = LAMc * av.w + hr * hp[3];
          Ar[kx] = av;
        }
      }
    } else if (sl == 0) {
      // final h -> output (every WG holds the full vector; slice 0 writes)
      #pragma unroll
      for (int e = 0; e < (16 * DHc) / BLK; e++) {
        int li = e * BLK + tid; int b = li / DHc, n = li - b * DHc;
        out[(size_t)(g * 16 + b) * DHc + n] = hs[b][n];
      }
    }
    __syncthreads();
  }
}

// ---------------- launch ----------------
extern "C" void kernel_launch(void* const* d_in, const int* in_sizes, int n_in,
                              void* d_out, int out_size, void* d_ws, size_t ws_size,
                              hipStream_t stream) {
  const float* z   = (const float*)d_in[0];   // [48,96,192]
  const float* Wh  = (const float*)d_in[1];   // [384,384]
  const float* Wg  = (const float*)d_in[2];   // [384,192]
  const float* bh  = (const float*)d_in[3];   // [384]
  const float* lng = (const float*)d_in[4];   // [384]
  const float* lnb = (const float*)d_in[5];   // [384]
  const float* al  = (const float*)d_in[6];   // [1]
  float* out = (float*)d_out;
  float* ws  = (float*)d_ws;

  const long nz = (long)WS_FLOATS;
  fw_rnn_zero<<<(unsigned)((nz + 255) / 256), 256, 0, stream>>>(ws, nz);

  const int npack = KT_H * NT * 512 + KT_G * NT * 512;
  fw_rnn_pack<<<(npack + 255) / 256, 256, 0, stream>>>(Wh, Wg, ws);

  fw_rnn_main<<<NWG, BLK, 0, stream>>>(z, bh, lng, lnb, al, out, ws);
}
